// MultiModalEncoder_69355131896070
// MI455X (gfx1250) — matrix operand order repaired and checked by reading, not compiled, not run on hardware
//
#include <hip/hip_runtime.h>

// ---------------------------------------------------------------------------
// MultiModalEncoder for MI455X (gfx1250, wave32, WMMA bf16)
//
// Math collapse (exact): all attentions are softmax over identical logits
// (seq len 1, duplicated rows) -> uniform 0.5 weights. Only Wv_* / Wp matter:
//   v_mv = mv @ Wv_mv + bv_mv ; v_pc = pc @ Wv_pc + bv_pc
//   r    = 0.5*(v_mv+v_pc) + max(v_mv, v_pc)
//   res  = 2*(r @ Wv_c + bv_c)
//   h    = res @ Wp + bp ; LayerNorm ; SELU ; h / (||h||+1e-9)
// ---------------------------------------------------------------------------

typedef __attribute__((ext_vector_type(16))) __bf16 v16bf;
typedef __attribute__((ext_vector_type(8)))  __bf16 v8bf;
typedef __attribute__((ext_vector_type(8)))  float  v8f;

#define BATCH      131072
#define MV_INW     256
#define PC_INW     128
#define DD         128
#define WAVES      8
#define BLOCK      (WAVES * 32)          // 256 threads
#define TILES      2                     // 16-row tiles per wave
#define ROWS_BLK   (WAVES * 16 * TILES)  // 256 rows per block
#define GRID       (BATCH / ROWS_BLK)    // 512 blocks

// LDS layout (bytes). Padded strides (bf16 elems) to avoid bank conflicts.
#define SMV_STRIDE  264                  // rows of WvT_mv (128 x 256)
#define S128_STRIDE 136                  // rows of 128-wide transposed weights
#define OFF_WVMV   0
#define OFF_WVPC   (OFF_WVMV + 128 * SMV_STRIDE * 2)   // 67584
#define OFF_WVC    (OFF_WVPC + 128 * S128_STRIDE * 2)  // 102400
#define OFF_WP     (OFF_WVC  + 128 * S128_STRIDE * 2)  // 137216
#define OFF_BIAS   (OFF_WP   + 128 * S128_STRIDE * 2)  // 172032 : 6*128 f32
#define OFF_STAGE  (OFF_BIAS + 6 * 128 * 4)            // 175104
#define STAGE_WAVE (2 * 16 * S128_STRIDE * 2)          // 8704 (rbuf + resbuf)
#define SMEM_BYTES (OFF_STAGE + WAVES * STAGE_WAVE)    // 244736 < 320K

// ---- helpers --------------------------------------------------------------

static __device__ __forceinline__ v8f wmma_bf16(v16bf a, v16bf b, v8f c) {
  // D = A(16x32 bf16) x B(32x16 bf16) + C(16x16 f32)
  return __builtin_amdgcn_wmma_f32_16x16x32_bf16(
      /*neg_a=*/false, a, /*neg_b=*/false, b,
      /*c_mod=*/(short)0, c, /*reuse_a=*/false, /*reuse_b=*/false);
}

// A-operand from LDS (row-major bf16): lane holds row, elems 0-7 at p,
// elems 8-15 at p+16 (K split per half-wave handled by caller's base offset).
static __device__ __forceinline__ v16bf lds_loadA(const __bf16* p) {
  v8bf lo = *(const v8bf*)(p);
  v8bf hi = *(const v8bf*)(p + 16);
  return __builtin_shufflevector(lo, hi, 0, 1, 2, 3, 4, 5, 6, 7,
                                 8, 9, 10, 11, 12, 13, 14, 15);
}

// B-operand from LDS (transposed weight, row = output column): elems 0-7 at p,
// elems 8-15 at p+16 (lanes 0-15: K 0-7,16-23; lanes 16-31: K 8-15,24-31 via base offs).
static __device__ __forceinline__ v16bf lds_loadB(const __bf16* p) {
  v8bf lo = *(const v8bf*)(p);
  v8bf hi = *(const v8bf*)(p + 16);
  return __builtin_shufflevector(lo, hi, 0, 1, 2, 3, 4, 5, 6, 7,
                                 8, 9, 10, 11, 12, 13, 14, 15);
}

// A-operand straight from global f32 row: elems 0-7 = p[0..7], 8-15 = p[16..23]
static __device__ __forceinline__ v16bf cvtA(const float* p) {
  float4 f0 = *(const float4*)(p);
  float4 f1 = *(const float4*)(p + 4);
  float4 f2 = *(const float4*)(p + 16);
  float4 f3 = *(const float4*)(p + 20);
  v16bf r;
  r[0] = (__bf16)f0.x;  r[1] = (__bf16)f0.y;  r[2] = (__bf16)f0.z;  r[3] = (__bf16)f0.w;
  r[4] = (__bf16)f1.x;  r[5] = (__bf16)f1.y;  r[6] = (__bf16)f1.z;  r[7] = (__bf16)f1.w;
  r[8] = (__bf16)f2.x;  r[9] = (__bf16)f2.y;  r[10] = (__bf16)f2.z; r[11] = (__bf16)f2.w;
  r[12] = (__bf16)f3.x; r[13] = (__bf16)f3.y; r[14] = (__bf16)f3.z; r[15] = (__bf16)f3.w;
  return r;
}

// sum across the 16-lane half-wave (C-layout rows never cross halves)
static __device__ __forceinline__ float red16(float x) {
  x += __shfl_xor(x, 1, 32);
  x += __shfl_xor(x, 2, 32);
  x += __shfl_xor(x, 4, 32);
  x += __shfl_xor(x, 8, 32);
  return x;
}

// ---- kernel ---------------------------------------------------------------

__global__ __launch_bounds__(BLOCK) void mm_encoder_kernel(
    const float* __restrict__ mv_enc, const float* __restrict__ pc_enc,
    const float* __restrict__ Wv_mv, const float* __restrict__ bv_mv,
    const float* __restrict__ Wv_pc, const float* __restrict__ bv_pc,
    const float* __restrict__ Wv_c,  const float* __restrict__ bv_c,
    const float* __restrict__ Wp,    const float* __restrict__ bp,
    const float* __restrict__ ln_g,  const float* __restrict__ ln_b,
    float* __restrict__ out) {
  extern __shared__ char smem[];
  __bf16* sWvMv = (__bf16*)(smem + OFF_WVMV);
  __bf16* sWvPc = (__bf16*)(smem + OFF_WVPC);
  __bf16* sWvC  = (__bf16*)(smem + OFF_WVC);
  __bf16* sWp   = (__bf16*)(smem + OFF_WP);
  float*  sBias = (float*)(smem + OFF_BIAS);

  const int tid = threadIdx.x;

  // Stage weights into LDS transposed + bf16 (coalesced global reads).
  for (int idx = tid; idx < MV_INW * 128; idx += BLOCK) {
    int k = idx >> 7, n = idx & 127;
    sWvMv[n * SMV_STRIDE + k] = (__bf16)Wv_mv[idx];
  }
  for (int idx = tid; idx < 128 * 128; idx += BLOCK) {
    int k = idx >> 7, n = idx & 127;
    sWvPc[n * S128_STRIDE + k] = (__bf16)Wv_pc[idx];
    sWvC [n * S128_STRIDE + k] = (__bf16)Wv_c [idx];
    sWp  [n * S128_STRIDE + k] = (__bf16)Wp  [idx];
  }
  if (tid < 128) {
    sBias[tid]       = bv_mv[tid];
    sBias[128 + tid] = bv_pc[tid];
    sBias[256 + tid] = bv_c[tid];
    sBias[384 + tid] = bp[tid];
    sBias[512 + tid] = ln_g[tid];
    sBias[640 + tid] = ln_b[tid];
  }
  __syncthreads();

  const int wave  = tid >> 5;
  const int lane  = tid & 31;
  const int l16   = lane & 15;
  const int lhalf = lane >> 4;

  __bf16* rb   = (__bf16*)(smem + OFF_STAGE + wave * STAGE_WAVE);
  __bf16* resb = rb + 16 * S128_STRIDE;

  for (int t = 0; t < TILES; ++t) {
    const int row0 = blockIdx.x * ROWS_BLK + (wave * TILES + t) * 16;

    // ---- load A tiles (16 rows) as bf16 ----
    const float* mvRow = mv_enc + (size_t)(row0 + l16) * MV_INW;
    const float* pcRow = pc_enc + (size_t)(row0 + l16) * PC_INW;
    v16bf aMv[8], aPc[4];
#pragma unroll
    for (int kc = 0; kc < 8; ++kc) aMv[kc] = cvtA(mvRow + kc * 32 + lhalf * 8);
#pragma unroll
    for (int kc = 0; kc < 4; ++kc) aPc[kc] = cvtA(pcRow + kc * 32 + lhalf * 8);

    // ---- GEMM1 (v_mv, v_pc) + combine -> r staged as bf16 in LDS ----
#pragma unroll
    for (int n = 0; n < 8; ++n) {
      v8f acc1, acc2;
      float b1 = sBias[n * 16 + l16];
      float b2 = sBias[128 + n * 16 + l16];
#pragma unroll
      for (int i = 0; i < 8; ++i) { acc1[i] = b1; acc2[i] = b2; }
      const __bf16* w1 = sWvMv + (n * 16 + l16) * SMV_STRIDE + lhalf * 8;
#pragma unroll
      for (int kc = 0; kc < 8; ++kc)
        acc1 = wmma_bf16(aMv[kc], lds_loadB(w1 + kc * 32), acc1);
      const __bf16* w2 = sWvPc + (n * 16 + l16) * S128_STRIDE + lhalf * 8;
#pragma unroll
      for (int kc = 0; kc < 4; ++kc)
        acc2 = wmma_bf16(aPc[kc], lds_loadB(w2 + kc * 32), acc2);
#pragma unroll
      for (int i = 0; i < 8; ++i) {
        float x = acc1[i], y = acc2[i];
        float r = 0.5f * (x + y) + fmaxf(x, y);
        rb[(i + 8 * lhalf) * S128_STRIDE + n * 16 + l16] = (__bf16)r;
      }
    }
    __syncthreads();

    // ---- GEMM2: res = 2*(r @ Wv_c + bv_c) -> staged ----
    v16bf aR[4];
    const __bf16* rrow = rb + l16 * S128_STRIDE + lhalf * 8;
#pragma unroll
    for (int kc = 0; kc < 4; ++kc) aR[kc] = lds_loadA(rrow + kc * 32);
#pragma unroll
    for (int n = 0; n < 8; ++n) {
      v8f acc;
      float b = sBias[256 + n * 16 + l16];
#pragma unroll
      for (int i = 0; i < 8; ++i) acc[i] = b;
      const __bf16* w = sWvC + (n * 16 + l16) * S128_STRIDE + lhalf * 8;
#pragma unroll
      for (int kc = 0; kc < 4; ++kc)
        acc = wmma_bf16(aR[kc], lds_loadB(w + kc * 32), acc);
#pragma unroll
      for (int i = 0; i < 8; ++i)
        resb[(i + 8 * lhalf) * S128_STRIDE + n * 16 + l16] = (__bf16)(2.0f * acc[i]);
    }
    __syncthreads();

    // ---- GEMM3: h = res @ Wp + bp (kept in registers, C layout) ----
    v16bf aRes[4];
    const __bf16* resrow = resb + l16 * S128_STRIDE + lhalf * 8;
#pragma unroll
    for (int kc = 0; kc < 4; ++kc) aRes[kc] = lds_loadA(resrow + kc * 32);
    v8f h[8];
#pragma unroll
    for (int n = 0; n < 8; ++n) {
      float b = sBias[384 + n * 16 + l16];
#pragma unroll
      for (int i = 0; i < 8; ++i) h[n][i] = b;
      const __bf16* w = sWp + (n * 16 + l16) * S128_STRIDE + lhalf * 8;
#pragma unroll
      for (int kc = 0; kc < 4; ++kc)
        h[n] = wmma_bf16(aRes[kc], lds_loadB(w + kc * 32), h[n]);
    }

    // ---- epilogue: LayerNorm -> SELU -> L2 normalize (per row) ----
#pragma unroll
    for (int i = 0; i < 8; ++i) {
      float s = 0.f;
#pragma unroll
      for (int n = 0; n < 8; ++n) s += h[n][i];
      float mu = red16(s) * 0.0078125f;  // /128
      float v = 0.f;
#pragma unroll
      for (int n = 0; n < 8; ++n) { float d = h[n][i] - mu; v += d * d; }
      v = red16(v) * 0.0078125f;
      float inv = rsqrtf(v + 1e-5f);
      float ss = 0.f;
#pragma unroll
      for (int n = 0; n < 8; ++n) {
        float g  = sBias[512 + n * 16 + l16];
        float bb = sBias[640 + n * 16 + l16];
        float y = (h[n][i] - mu) * inv * g + bb;
        y = (y > 0.f) ? (1.0507009873554805f * y)
                      : (1.0507009873554805f * 1.6732632423543772f * (__expf(y) - 1.f));
        h[n][i] = y;
        ss += y * y;
      }
      ss = red16(ss);
      float invn = 1.f / (sqrtf(ss) + 1e-9f);
#pragma unroll
      for (int n = 0; n < 8; ++n) h[n][i] *= invn;
    }

    // ---- store (f32, row-major) ----
#pragma unroll
    for (int i = 0; i < 8; ++i) {
      size_t r = (size_t)(row0 + i + 8 * lhalf);
#pragma unroll
      for (int n = 0; n < 8; ++n)
        out[r * DD + n * 16 + l16] = h[n][i];
    }
    __syncthreads();  // protect per-wave staging reuse across tiles
  }
}

// ---------------------------------------------------------------------------

extern "C" void kernel_launch(void* const* d_in, const int* in_sizes, int n_in,
                              void* d_out, int out_size, void* d_ws, size_t ws_size,
                              hipStream_t stream) {
  (void)in_sizes; (void)n_in; (void)out_size; (void)d_ws; (void)ws_size;
  const float* mv_enc = (const float*)d_in[0];
  const float* pc_enc = (const float*)d_in[1];
  const float* Wv_mv  = (const float*)d_in[6];
  const float* bv_mv  = (const float*)d_in[7];
  const float* Wv_pc  = (const float*)d_in[12];
  const float* bv_pc  = (const float*)d_in[13];
  const float* Wv_c   = (const float*)d_in[18];
  const float* bv_c   = (const float*)d_in[19];
  const float* Wp     = (const float*)d_in[20];
  const float* bp     = (const float*)d_in[21];
  const float* ln_g   = (const float*)d_in[22];
  const float* ln_b   = (const float*)d_in[23];
  float* out = (float*)d_out;

  (void)hipFuncSetAttribute((const void*)mm_encoder_kernel,
                            hipFuncAttributeMaxDynamicSharedMemorySize,
                            SMEM_BYTES);
  mm_encoder_kernel<<<GRID, BLOCK, SMEM_BYTES, stream>>>(
      mv_enc, pc_enc, Wv_mv, bv_mv, Wv_pc, bv_pc, Wv_c, bv_c, Wp, bp,
      ln_g, ln_b, out);
}
